// MultiChannelS4D_18202071400696
// MI455X (gfx1250) — matrix operand order repaired and checked by reading, not compiled, not run on hardware
//
#include <hip/hip_runtime.h>
#include <hip/hip_bf16.h>
#include <math.h>
#include <stdint.h>

typedef __attribute__((ext_vector_type(16))) _Float16 v16h;
typedef __attribute__((ext_vector_type(8)))  _Float16 v8h;
typedef __attribute__((ext_vector_type(8)))  float    v8f;
typedef __attribute__((ext_vector_type(4)))  int      v4i;

#define T_LEN 2048
#define K_CH  8
#define B_SZ  8
#define D_DIM 64
#define M_PAD 64          // zero pad in front of M in LDS (covers causal under-run)
#define HT_STRIDE 40      // halves per transposed-h row (32 data + 8 pad, 16B align)
#define CHUNK_HALVES 2048 // 64 d-rows * 32 s-cols per chunk in ws
#define BUF_HALVES (D_DIM * HT_STRIDE)   // 2560 halves per LDS chunk buffer

#if defined(__has_builtin)
#if __has_builtin(__builtin_amdgcn_global_load_async_to_lds_b128) && \
    __has_builtin(__builtin_amdgcn_s_wait_asynccnt)
#define USE_ASYNC 1
#endif
#endif
#ifndef USE_ASYNC
#define USE_ASYNC 0
#endif

#if USE_ASYNC
typedef __attribute__((address_space(1))) v4i as1_v4i;
typedef __attribute__((address_space(3))) v4i as3_v4i;
#endif

// ---------------------------------------------------------------------------
// Kernel 1: M[k,t] = 2*Re( exp(eig_k*dt*t) * Bbar_k ), stored f16 in ws.
// ---------------------------------------------------------------------------
__global__ void s4d_coeff_kernel(const float* __restrict__ log_neg_re,
                                 const float* __restrict__ imag,
                                 const float* __restrict__ B_proj,
                                 const float* __restrict__ log_dt,
                                 _Float16* __restrict__ Mws) {
  int tid = blockIdx.x * blockDim.x + threadIdx.x;
  if (tid >= K_CH * T_LEN) return;
  int k = tid >> 11;
  int t = tid & (T_LEN - 1);

  float re = -expf(log_neg_re[k]);
  float im = imag[k];
  float dt = expf(log_dt[0]);

  float ed = expf(re * dt);
  float ar = ed * cosf(im * dt);
  float ai = ed * sinf(im * dt);
  float nr = ar - 1.0f, ni = ai;
  float den = re * re + im * im;
  float bp  = B_proj[k];
  float br = (nr * re + ni * im) / den * bp;
  float bi = (ni * re - nr * im) / den * bp;
  float s = dt * (float)t;
  float pm = expf(re * s);
  float pr = pm * cosf(im * s);
  float pi = pm * sinf(im * s);

  Mws[k * T_LEN + t] = (_Float16)(2.0f * (pr * br - pi * bi));
}

// ---------------------------------------------------------------------------
// Kernel 2: pre-convert h to f16, transposed per 32-step chunk:
//   hT16[((b*64 + c)*64 + d)*32 + s] = (f16) h[b, c*32+s, d]
// One block per (b, chunk): read coalesced f32, transpose via LDS, write b128.
// ---------------------------------------------------------------------------
__global__ void __launch_bounds__(256)
h_pack_kernel(const float* __restrict__ h, _Float16* __restrict__ hT16) {
  __shared__ _Float16 tile[D_DIM * HT_STRIDE];
  const int b = blockIdx.x >> 6;
  const int c = blockIdx.x & 63;
  const int tid = threadIdx.x;
  const int d = tid & 63;
  const int srow = tid >> 6;                      // 0..3, 8 passes
  const float* src = h + ((size_t)b * T_LEN + (size_t)c * 32 + srow) * D_DIM + d;
#pragma unroll
  for (int j = 0; j < 8; ++j)
    tile[d * HT_STRIDE + srow + 4 * j] = (_Float16)src[(size_t)(4 * j) * D_DIM];
  __syncthreads();
  // thread writes 8 contiguous halves: i0 = tid*8 -> d = tid>>2, s = (tid&3)*8
  v8h v = *(const v8h*)&tile[(tid >> 2) * HT_STRIDE + (tid & 3) * 8];
  *(v8h*)&hT16[(size_t)(b * 64 + c) * CHUNK_HALVES + (size_t)tid * 8] = v;
}

// ---------------------------------------------------------------------------
// Chunk staging: copy one packed 4KB chunk (64 x 32 halves) from ws into a
// padded LDS buffer.  Async (ASYNCcnt) path when available, sync fallback.
// 128 threads, 2 b128 transfers each.
// ---------------------------------------------------------------------------
__device__ __forceinline__ void stage_chunk(const _Float16* __restrict__ gsrc,
                                            _Float16* lbuf, int tid) {
  const int i0 = tid * 2;
#pragma unroll
  for (int j = 0; j < 2; ++j) {
    const int i = i0 + j;
    const int dst = (i >> 2) * HT_STRIDE + (i & 3) * 8;  // d = i>>2, s = (i&3)*8
#if USE_ASYNC
    as1_v4i* g = (as1_v4i*)(uintptr_t)(gsrc + (size_t)i * 8);
    as3_v4i* l = (as3_v4i*)(uint32_t)(uintptr_t)(lbuf + dst);
    __builtin_amdgcn_global_load_async_to_lds_b128(g, l, 0, 0);
#else
    *(v8h*)(lbuf + dst) = *(const v8h*)(gsrc + (size_t)i * 8);
#endif
  }
}

// ---------------------------------------------------------------------------
// Kernel 3: out[b,t,k,d] = sum_{s<=t} M[t-s,k] * h[b,s,d]
// Block = 128 threads (4 waves), one (b,k), 256-row tile.
// Wave tile: 64 rows x 64 cols -> 4x4 f32 accumulators, 16 WMMA per chunk.
// Double-buffered async LDS staging of packed h chunks.
// ---------------------------------------------------------------------------
__global__ void __launch_bounds__(128)
s4d_wmma_kernel(const _Float16* __restrict__ hT16,
                const _Float16* __restrict__ Mws,
                float* __restrict__ out) {
  __shared__ _Float16 Mlds[M_PAD + T_LEN];        // 4224 B
  __shared__ _Float16 hbuf[2][BUF_HALVES];        // 2 x 5120 B

  const int bid  = blockIdx.x;
  const int tile = bid & 7;            // fastest-varying: balances chunk counts
  const int k    = (bid >> 3) & 7;
  const int b    = bid >> 6;
  const int tid  = threadIdx.x;
  const int lane = tid & 31;
  const int wave = tid >> 5;

  // Stage M channel k with leading zero pad (causal mask becomes data).
  for (int i = tid; i < M_PAD + T_LEN; i += 128)
    Mlds[i] = (i < M_PAD) ? (_Float16)0.0f : Mws[k * T_LEN + (i - M_PAD)];

  const int tile_base = tile * 256;
  const int t0        = tile_base + wave * 64;    // wave's first output row
  const int cmax      = tile * 8 + 7;             // last chunk for this tile

  v8f acc[4][4] = {};

  const int n       = lane & 15;       // B col / output col within 16-tile
  const int halfsel = lane >> 4;
  const int off8    = halfsel << 3;    // A-frag K offset
  const int cOff    = halfsel << 3;    // B-frag K offset

  const _Float16* __restrict__ hchunks = hT16 + (size_t)b * 64 * CHUNK_HALVES;

  // Prologue: stage chunk 0 into buffer 0.
  stage_chunk(hchunks, hbuf[0], tid);

  for (int c = 0; c <= cmax; ++c) {
    const _Float16* cur = hbuf[c & 1];
    if (c < cmax) {
      stage_chunk(hchunks + (size_t)(c + 1) * CHUNK_HALVES, hbuf[(c + 1) & 1], tid);
#if USE_ASYNC
      __builtin_amdgcn_s_wait_asynccnt(2);   // drain chunk c, keep c+1 in flight
#endif
    } else {
#if USE_ASYNC
      __builtin_amdgcn_s_wait_asynccnt(0);
#endif
    }
    __syncthreads();                          // all waves' chunk-c data visible

    const int s0 = c << 5;
    if (s0 <= t0 + 63) {                      // causal; wave-uniform -> EXEC full
      // A fragments: Toeplitz gather.  ISA A 16x32 f16 layout:
      // row m = lane%16, elem e -> K = (e<8?0:16) + (lane<16?0:8) + (e&7)
      v16h a[4];
      const int q0 = M_PAD + t0 + (lane & 15) - off8 - s0;
#pragma unroll
      for (int r = 0; r < 4; ++r) {
#pragma unroll
        for (int e = 0; e < 16; ++e) {
          int dk = (e & 7) + ((e & 8) << 1);
          a[r][e] = Mlds[q0 + 16 * r - dk];
        }
      }
#pragma unroll
      for (int c4 = 0; c4 < 4; ++c4) {
        // B fragment: 32x16, col n = lane%16, elem e -> K = (e<8?0:16) + cOff + (e&7)
        const _Float16* hp = cur + (c4 * 16 + n) * HT_STRIDE + cOff;
        v8h blo = *(const v8h*)hp;
        v8h bhi = *(const v8h*)(hp + 16);
        v16h bf = __builtin_shufflevector(blo, bhi,
                    0, 1, 2, 3, 4, 5, 6, 7, 8, 9, 10, 11, 12, 13, 14, 15);
#pragma unroll
        for (int r = 0; r < 4; ++r)
          acc[r][c4] = __builtin_amdgcn_wmma_f32_16x16x32_f16(
              false, a[r], false, bf, (short)0, acc[r][c4], false, false);
      }
    }
    __syncthreads();                          // reads done before buffer reuse
  }

  // Store. C/D layout: VGPR r -> row r + 8*halfsel, col = lane%16.
#pragma unroll
  for (int r2 = 0; r2 < 4; ++r2) {
#pragma unroll
    for (int c4 = 0; c4 < 4; ++c4) {
#pragma unroll
      for (int r = 0; r < 8; ++r) {
        int row = t0 + r2 * 16 + r + (halfsel << 3);
        int col = c4 * 16 + n;
        out[(((size_t)b * T_LEN + row) * K_CH + k) * D_DIM + col] = acc[r2][c4][r];
      }
    }
  }
}

// ---------------------------------------------------------------------------
extern "C" void kernel_launch(void* const* d_in, const int* in_sizes, int n_in,
                              void* d_out, int out_size, void* d_ws, size_t ws_size,
                              hipStream_t stream) {
  const float* h          = (const float*)d_in[0];
  const float* log_neg_re = (const float*)d_in[1];
  const float* imag       = (const float*)d_in[2];
  const float* B_proj     = (const float*)d_in[3];
  const float* log_dt     = (const float*)d_in[4];

  // ws layout: [0, 2MB): packed f16 transposed h ; [2MB, 2MB+32KB): M table
  _Float16* hT16 = (_Float16*)d_ws;
  _Float16* Mws  = (_Float16*)((char*)d_ws +
                               (size_t)B_SZ * T_LEN * D_DIM * sizeof(_Float16));
  float* out = (float*)d_out;

  s4d_coeff_kernel<<<(K_CH * T_LEN + 255) / 256, 256, 0, stream>>>(
      log_neg_re, imag, B_proj, log_dt, Mws);

  h_pack_kernel<<<B_SZ * (T_LEN / 32), 256, 0, stream>>>(h, hT16);

  const int nblocks = (T_LEN / 256) * K_CH * B_SZ;   // 8 * 8 * 8 = 512
  s4d_wmma_kernel<<<nblocks, 128, 0, stream>>>(hT16, Mws, out);
}
